// kerl_17368847745534
// MI455X (gfx1250) — matrix operand order repaired and checked by reading, not compiled, not run on hardware
//
#include <hip/hip_runtime.h>
#include <hip/hip_bf16.h>

typedef __attribute__((ext_vector_type(16))) _Float16 v16h;
typedef __attribute__((ext_vector_type(8)))  float    v8f;
typedef _Float16 half_t;

#define BB 1024
#define LL 50
#define DD 50
#define VV 100000
#define KPAD 64
#define MTILES 64      // 1024 / 16
#define NTILES 6250    // 100000 / 16
#define NTG 3125       // NTILES / 2 (two N-tiles per wave)

// ---------------- Kernel 1: per-column mean / inv-std of kg_map ----------------
__global__ void kg_stats_kernel(const float* __restrict__ kg,
                                float* __restrict__ mu, float* __restrict__ istd) {
    int d = blockIdx.x;                 // 0..49
    float s = 0.f, s2 = 0.f;
    for (int i = threadIdx.x; i < VV; i += blockDim.x) {
        float v = kg[i * DD + d];
        s += v; s2 += v * v;
    }
    __shared__ float sh[256], sh2[256];
    sh[threadIdx.x] = s; sh2[threadIdx.x] = s2;
    __syncthreads();
    for (int off = 128; off > 0; off >>= 1) {
        if ((int)threadIdx.x < off) {
            sh[threadIdx.x]  += sh[threadIdx.x + off];
            sh2[threadIdx.x] += sh2[threadIdx.x + off];
        }
        __syncthreads();
    }
    if (threadIdx.x == 0) {
        float m   = sh[0] / (float)VV;
        float var = sh2[0] / (float)VV - m * m;
        mu[d]   = m;
        istd[d] = rsqrtf(var + 1e-5f);
    }
}

// ---------------- Kernel 2: batch_kg = (sum_t kg[seq] - L*mu) * istd / len ----------------
__global__ void batch_kg_kernel(const int* __restrict__ seq, const int* __restrict__ tlen,
                                const float* __restrict__ kg,
                                const float* __restrict__ mu, const float* __restrict__ istd,
                                float* __restrict__ bkg) {
    int idx = blockIdx.x * blockDim.x + threadIdx.x;
    if (idx >= BB * DD) return;
    int b = idx / DD, d = idx % DD;
    float s = 0.f;
    for (int t = 0; t < LL; ++t) s += kg[seq[b * LL + t] * DD + d];
    bkg[idx] = (s - (float)LL * mu[d]) * istd[d] / (float)tlen[b];
}

// ---------------- Kernel 3: GRU, one workgroup per sample, lanes = hidden dims ----------------
__global__ void gru_kernel(const int* __restrict__ seq, const int* __restrict__ tlen,
                           const float* __restrict__ emb,
                           const float* __restrict__ Wih, const float* __restrict__ Whh,
                           const float* __restrict__ bih, const float* __restrict__ bhh,
                           float* __restrict__ hout) {
    int b = blockIdx.x;
    int d = threadIdx.x;                // 0..63, dims 0..49 active
    __shared__ float xs[DD];
    __shared__ float hs[DD];
    if (d < DD) hs[d] = 0.f;
    int len = tlen[b];
    for (int t = 0; t < LL; ++t) {
        __syncthreads();
        if (d < DD) {
            int tok = seq[b * LL + t];
            xs[d] = emb[tok * DD + d];
        }
        __syncthreads();
        float hnew = 0.f;
        if (d < DD) {
            float ir = bih[d], iz = bih[DD + d], in_ = bih[2 * DD + d];
            float hr = bhh[d], hz = bhh[DD + d], hn_ = bhh[2 * DD + d];
            const float* wir = Wih + d * DD;
            const float* wiz = Wih + (DD + d) * DD;
            const float* win = Wih + (2 * DD + d) * DD;
            const float* whr = Whh + d * DD;
            const float* whz = Whh + (DD + d) * DD;
            const float* whn = Whh + (2 * DD + d) * DD;
            #pragma unroll
            for (int j = 0; j < DD; ++j) {
                float xv = xs[j], hv = hs[j];
                ir  += wir[j] * xv;  iz  += wiz[j] * xv;  in_ += win[j] * xv;
                hr  += whr[j] * hv;  hz  += whz[j] * hv;  hn_ += whn[j] * hv;
            }
            float r = 1.f / (1.f + expf(-(ir + hr)));
            float z = 1.f / (1.f + expf(-(iz + hz)));
            float n = tanhf(in_ + r * hn_);
            hnew = (1.f - z) * n + z * hs[d];
        }
        __syncthreads();
        if (d < DD) {
            hs[d] = hnew;
            if (t == len - 1) hout[b * DD + d] = hnew;
        }
    }
}

// ---------------- Kernel 4: hidden = tanh([h, bkg] @ mlp_W.T + mlp_b) ----------------
__global__ void mlp_kernel(const float* __restrict__ hin, const float* __restrict__ bkg,
                           const float* __restrict__ mlpW, const float* __restrict__ mlpb,
                           float* __restrict__ hidden) {
    int idx = blockIdx.x * blockDim.x + threadIdx.x;
    if (idx >= BB * DD) return;
    int b = idx / DD, d = idx % DD;
    const float* w = mlpW + d * (2 * DD);
    float acc = mlpb[d];
    #pragma unroll
    for (int j = 0; j < DD; ++j) acc += w[j] * hin[b * DD + j];
    #pragma unroll
    for (int j = 0; j < DD; ++j) acc += w[DD + j] * bkg[b * DD + j];
    hidden[idx] = tanhf(acc);
}

// ---------------- Kernel 5: pack A (hidden) into WMMA 16-bit A-fragment layout ----------------
// layout index: (((mt*2 + kb)*32 + lane)*16 + j) ; k = kb*32 + j + (lane>=16?16:0)
__global__ void pack_a_kernel(const float* __restrict__ hidden, half_t* __restrict__ apack) {
    int idx = blockIdx.x * blockDim.x + threadIdx.x;
    if (idx >= MTILES * 2 * 32 * 16) return;
    int j    = idx & 15;
    int lane = (idx >> 4) & 31;
    int kb   = (idx >> 9) & 1;
    int mt   = idx >> 10;
    int m = mt * 16 + (lane & 15);
    int k = kb * 32 + j + ((lane >= 16) ? 16 : 0);
    float v = (k < DD) ? hidden[m * DD + k] : 0.f;
    apack[idx] = (half_t)v;
}

// ---------------- Kernel 6: fc_W f32 [V,50] -> f16 row-major padded [V,64] ----------------
__global__ void pack_b_kernel(const float* __restrict__ fcW, half_t* __restrict__ fcwh) {
    int idx = blockIdx.x * blockDim.x + threadIdx.x;
    if (idx >= VV * KPAD) return;
    int k = idx & (KPAD - 1);
    int row = idx >> 6;
    fcwh[idx] = (half_t)((k < DD) ? fcW[row * DD + k] : 0.f);
}

// ---------------- Kernel 7: FC GEMM via v_wmma_f32_16x16x32_f16 ----------------
// Each wave: one 16-row M-tile x two 16-col N-tiles; K = 64 = 2 WMMA k-steps.
__global__ void fc_wmma_kernel(const half_t* __restrict__ apack,
                               const half_t* __restrict__ fcwh,
                               const float* __restrict__ fcb,
                               float* __restrict__ out) {
    int wave = (blockIdx.x * blockDim.x + threadIdx.x) >> 5;   // 0 .. 199999
    int lane = threadIdx.x & 31;
    int mt   = wave / NTG;
    int ntg  = wave % NTG;

    // A fragments for both k-blocks (contiguous 32B per lane)
    v16h a0 = *(const v16h*)(apack + (((size_t)mt * 2 + 0) * 32 + lane) * 16);
    v16h a1 = *(const v16h*)(apack + (((size_t)mt * 2 + 1) * 32 + lane) * 16);

    int m0   = mt * 16 + ((lane >= 16) ? 8 : 0);
    int koff = (lane < 16) ? 0 : 16;

    #pragma unroll
    for (int q = 0; q < 2; ++q) {
        int nt  = ntg * 2 + q;
        int col = nt * 16 + (lane & 15);          // this lane's output column (= fc_W row)
        const half_t* brow = fcwh + (size_t)col * KPAD;
        v16h b0 = *(const v16h*)(brow + koff);        // k 0..31 block
        v16h b1 = *(const v16h*)(brow + 32 + koff);   // k 32..63 block
        v8f c = {};
        c = __builtin_amdgcn_wmma_f32_16x16x32_f16(false, a0, false, b0, (short)0, c, false, false);
        c = __builtin_amdgcn_wmma_f32_16x16x32_f16(false, a1, false, b1, (short)0, c, false, false);
        float bias = fcb[col];
        #pragma unroll
        for (int r = 0; r < 8; ++r) {
            out[(size_t)(m0 + r) * VV + col] = c[r] + bias;
        }
    }
}

extern "C" void kernel_launch(void* const* d_in, const int* in_sizes, int n_in,
                              void* d_out, int out_size, void* d_ws, size_t ws_size,
                              hipStream_t stream) {
    const int*   seq  = (const int*)d_in[0];
    const int*   tlen = (const int*)d_in[1];
    const float* emb  = (const float*)d_in[2];
    const float* kg   = (const float*)d_in[3];
    const float* Wih  = (const float*)d_in[4];
    const float* Whh  = (const float*)d_in[5];
    const float* bih  = (const float*)d_in[6];
    const float* bhh  = (const float*)d_in[7];
    const float* mlpW = (const float*)d_in[8];
    const float* mlpb = (const float*)d_in[9];
    const float* fcW  = (const float*)d_in[10];
    const float* fcb  = (const float*)d_in[11];
    float* out = (float*)d_out;

    char* ws = (char*)d_ws;
    size_t off = 0;
    auto alloc = [&](size_t bytes) -> char* {
        char* p = ws + off;
        off = (off + bytes + 255) & ~(size_t)255;
        return p;
    };
    float*  mu     = (float*)alloc(DD * sizeof(float));
    float*  istd   = (float*)alloc(DD * sizeof(float));
    float*  bkg    = (float*)alloc((size_t)BB * DD * sizeof(float));
    float*  hout   = (float*)alloc((size_t)BB * DD * sizeof(float));
    float*  hidden = (float*)alloc((size_t)BB * DD * sizeof(float));
    half_t* apack  = (half_t*)alloc((size_t)MTILES * 2 * 32 * 16 * sizeof(half_t));
    half_t* fcwh   = (half_t*)alloc((size_t)VV * KPAD * sizeof(half_t));

    kg_stats_kernel<<<DD, 256, 0, stream>>>(kg, mu, istd);
    batch_kg_kernel<<<(BB * DD + 255) / 256, 256, 0, stream>>>(seq, tlen, kg, mu, istd, bkg);
    gru_kernel<<<BB, 64, 0, stream>>>(seq, tlen, emb, Wih, Whh, bih, bhh, hout);
    mlp_kernel<<<(BB * DD + 255) / 256, 256, 0, stream>>>(hout, bkg, mlpW, mlpb, hidden);
    pack_a_kernel<<<(MTILES * 2 * 32 * 16 + 255) / 256, 256, 0, stream>>>(hidden, apack);
    pack_b_kernel<<<((size_t)VV * KPAD + 255) / 256, 256, 0, stream>>>(fcW, fcwh);
    // 64 Mtiles * 3125 wave-groups = 200000 waves = 25000 blocks of 8 waves, no divergence
    fc_wmma_kernel<<<(MTILES * NTG) / 8, 256, 0, stream>>>(apack, fcwh, fcb, out);
}
